// SSIMLoss_85598698209786
// MI455X (gfx1250) — matrix operand order repaired and checked by reading, not compiled, not run on hardware
//
#include <hip/hip_runtime.h>
#include <cstdint>

// MI455X / gfx1250, wave32. SSIM loss via WMMA-based separable Gaussian blur.
//
// Per 16x16 tile and per quantity q in {x, y, x^2, y^2, xy}:
//   Hlo = A(rows 0-15)  x Bh      (v_wmma_f32_16x16x32_f16)
//   Hhi = A(rows 16-31) x Bh
//   At  = per-lane pack of cvt_f16(Hlo), cvt_f16(Hhi)   == H^T in A layout
//   O^T = At x Bh                 (same band matrix: Wv^T == Bh)
// The D->A relayout is lane-local (no shuffles): D-layout lane n+16*(M>=8),
// VGPR M%8 composes exactly with A-layout K = 16*(h>>3) + 8*hi + (h&7).
// All five outputs are transposed identically; SSIM map + mean don't care.

typedef __attribute__((ext_vector_type(16))) _Float16 v16h;
typedef __attribute__((ext_vector_type(8)))  float    v8f;

// Normalized 11-tap Gaussian, sigma=1.5 (matches jnp reference to ~1e-7).
__constant__ float GW[11] = {
    0.00102837f, 0.00759876f, 0.03600078f, 0.10936069f, 0.21300554f,
    0.26601172f,
    0.21300554f, 0.10936069f, 0.03600078f, 0.00759876f, 0.00102837f
};

union V16H { uint32_t u[8]; v16h v; _Float16 h[16]; };

#define LROWS 32
#define LCOLS 144   // 128 output cols + 2*5 halo + 6 slack (A needs 32-col windows)
#define LSTR  146   // padded half-stride (73 dwords) -> conflict-free gathers

__global__ __launch_bounds__(256) void ssim_main(
    const float* __restrict__ X, const float* __restrict__ Y,
    double* __restrict__ partial)
{
    __shared__ _Float16 sx[LROWS * LSTR];
    __shared__ _Float16 sy[LROWS * LSTR];
    __shared__ _Float16 gtab[48];      // zero-padded band weights, index d+15
    __shared__ double   wsum[8];

    const int tid = threadIdx.x;
    const int gr0 = (int)blockIdx.y * 16  - 5;   // patch row 0 -> global row
    const int gc0 = (int)blockIdx.x * 128 - 5;   // patch col 0 -> global col
    const size_t planeOff = (size_t)blockIdx.z * (512u * 512u);

    // Band-weight table: gtab[15+d] = g[d] for d in [0,10], else 0.
    if (tid < 48) {
        float gv = (tid >= 15 && tid < 26) ? GW[tid - 15] : 0.f;
        gtab[tid] = (_Float16)gv;
    }

    // ---- Stage 32x144 halo patch (zero-padded) of x and y as f16 in LDS ----
    #pragma unroll
    for (int it = 0; it < 18; ++it) {            // 32*144 / 256 = 18
        int idx = tid + it * 256;
        int pr = idx / LCOLS;
        int pc = idx - pr * LCOLS;
        int gr = gr0 + pr, gc = gc0 + pc;
        float vx = 0.f, vy = 0.f;
        if ((unsigned)gr < 512u && (unsigned)gc < 512u) {
            size_t o = planeOff + (size_t)gr * 512u + (size_t)gc;
            vx = X[o]; vy = Y[o];
        }
        sx[pr * LSTR + pc] = (_Float16)vx;
        sy[pr * LSTR + pc] = (_Float16)vy;
    }
    __syncthreads();

    const int lane = tid & 31;
    const int w    = tid >> 5;       // wave id: 16-col tile within the strip
    const int m    = lane & 15;
    const int hi   = lane >> 4;      // 0: lanes 0-15, 1: lanes 16-31

    // ---- Gather A operands (16-bit A 16x32 ISA layout) for x and y ----
    // dword j of a lane holds K = (j<4?0:16) + 8*hi + 2*(j&3), K+1
    const uint32_t* sxu = (const uint32_t*)sx;
    const uint32_t* syu = (const uint32_t*)sy;
    V16H ax[2], ay[2];
    #pragma unroll
    for (int t = 0; t < 2; ++t) {                 // row halves: rows 0-15 / 16-31
        int base = (16 * t + m) * LSTR + 16 * w;  // halfs
        #pragma unroll
        for (int j = 0; j < 8; ++j) {
            int kj = ((j < 4) ? 0 : 16) + 8 * hi + 2 * (j & 3);
            ax[t].u[j] = sxu[(base + kj) >> 1];
            ay[t].u[j] = syu[(base + kj) >> 1];
        }
    }

    // ---- Band matrix Bh (B 32x16 ISA layout): lane col n=m, row K=8*hi+j (j<8) / 16+8*hi+(j-8) (j>=8),
    //      Bh[K][n] = g[K-n]. Used by BOTH passes (Wv^T == Bh). ----
    V16H bh;
    const int gbase = 15 + 8 * hi - m;
    #pragma unroll
    for (int j = 0; j < 16; ++j)
        bh.h[j] = gtab[gbase + j + ((j < 8) ? 0 : 8)];

    const v8f zc = {0.f, 0.f, 0.f, 0.f, 0.f, 0.f, 0.f, 0.f};

    // Separable blur: 2 horizontal WMMAs, lane-local repack, 1 vertical WMMA.
    auto blur = [&](v16h a0, v16h a1) -> v8f {
        v8f hlo = __builtin_amdgcn_wmma_f32_16x16x32_f16(
            false, a0, false, bh.v, (short)0, zc, false, false);
        v8f hhi = __builtin_amdgcn_wmma_f32_16x16x32_f16(
            false, a1, false, bh.v, (short)0, zc, false, false);
        V16H at;                      // H^T in A layout: no cross-lane traffic
        #pragma unroll
        for (int v = 0; v < 8; ++v) {
            at.h[v]     = (_Float16)hlo[v];
            at.h[v + 8] = (_Float16)hhi[v];
        }
        return __builtin_amdgcn_wmma_f32_16x16x32_f16(
            false, at.v, false, bh.v, (short)0, zc, false, false);
    };

    v8f ox  = blur(ax[0].v, ax[1].v);
    v8f oy  = blur(ay[0].v, ay[1].v);
    v8f oxx = blur(ax[0].v * ax[0].v, ax[1].v * ax[1].v);  // v_pk_mul_f16
    v8f oyy = blur(ay[0].v * ay[0].v, ay[1].v * ay[1].v);
    v8f oxy = blur(ax[0].v * ay[0].v, ax[1].v * ay[1].v);

    // ---- SSIM map (elementwise on transposed D tiles) + per-lane sum ----
    const float C1 = 1e-4f, C2 = 9e-4f;
    float s = 0.f;
    #pragma unroll
    for (int e = 0; e < 8; ++e) {
        float mx = ox[e], my = oy[e];
        float vxx = oxx[e] - mx * mx;
        float vyy = oyy[e] - my * my;
        float vxy = oxy[e] - mx * my;
        float num = (2.f * mx * my + C1) * (2.f * vxy + C2);
        float den = (mx * mx + my * my + C1) * (vxx + vyy + C2);
        s += num / (den + 1e-8f);
    }

    // ---- Deterministic block partial ----
    #pragma unroll
    for (int off = 16; off > 0; off >>= 1) s += __shfl_down(s, off, 32);
    if (lane == 0) wsum[w] = (double)s;
    __syncthreads();
    if (tid == 0) {
        double t = 0.0;
        #pragma unroll
        for (int i = 0; i < 8; ++i) t += wsum[i];
        partial[((size_t)blockIdx.z * gridDim.y + blockIdx.y) * gridDim.x
                + blockIdx.x] = t;
    }
}

__global__ __launch_bounds__(256) void ssim_reduce(
    const double* __restrict__ partial, float* __restrict__ out, int n)
{
    __shared__ double sm[256];
    double t = 0.0;
    for (int i = threadIdx.x; i < n; i += 256) t += partial[i];
    sm[threadIdx.x] = t;
    __syncthreads();
    #pragma unroll
    for (int off = 128; off > 0; off >>= 1) {
        if ((int)threadIdx.x < off) sm[threadIdx.x] += sm[threadIdx.x + off];
        __syncthreads();
    }
    if (threadIdx.x == 0)
        out[0] = 1.0f - (float)(sm[0] / 12582912.0);   // N = 16*3*512*512
}

extern "C" void kernel_launch(void* const* d_in, const int* in_sizes, int n_in,
                              void* d_out, int out_size, void* d_ws, size_t ws_size,
                              hipStream_t stream) {
    const float* x = (const float*)d_in[0];
    const float* y = (const float*)d_in[1];
    double* partial = (double*)d_ws;        // 4*32*48 = 6144 doubles (48 KiB)

    dim3 grid(4, 32, 48);                   // col strips, row strips, N*C planes
    ssim_main<<<grid, 256, 0, stream>>>(x, y, partial);
    ssim_reduce<<<1, 256, 0, stream>>>(partial, (float*)d_out, 4 * 32 * 48);
}
